// TransLayer_61795989455401
// MI455X (gfx1250) — matrix operand order repaired and checked by reading, not compiled, not run on hardware
//
#include <hip/hip_runtime.h>

// ---------------------------------------------------------------------------
// Nystromformer block for MI455X (gfx1250, wave32, WMMA bf16 -> f32 accum)
// ---------------------------------------------------------------------------
// B=4, N=4096, DIM=512, H=8, DH=64, M(landmarks)=256, l=16, pinv iters=6.
// All matmuls run on v_wmma_f32_16x16x32_bf16. a3-softmax is fused flash-style
// with the @v product (P staged via LDS). (a1@a2inv)@(a3@v) re-associated to
// a1 @ (a2inv @ (a3@v)). The two large square GEMMs (QKV, out-proj) use the
// CDNA5 async path: GLOBAL_LOAD_ASYNC_TO_LDS_B128 double-buffered tile staging
// ordered with s_wait_asynccnt + barriers.
// ---------------------------------------------------------------------------

typedef __attribute__((ext_vector_type(16))) __bf16 v16bf;
typedef __attribute__((ext_vector_type(8)))  float  v8f;

union FragBF { v16bf v; unsigned u[8]; unsigned short h[16]; };

__device__ inline float bf2f(unsigned short h) {
  unsigned u = ((unsigned)h) << 16; float f; __builtin_memcpy(&f, &u, 4); return f;
}
__device__ inline unsigned short f2bf(float f) {
  unsigned u; __builtin_memcpy(&u, &f, 4);
  unsigned r = u + 0x7FFFu + ((u >> 16) & 1u);      // round-to-nearest-even
  return (unsigned short)(r >> 16);
}
__device__ inline v8f vzero() {
  v8f c;
#pragma unroll
  for (int i = 0; i < 8; ++i) c[i] = 0.f;
  return c;
}
__device__ inline v8f wmma_bf16(v16bf a, v16bf b, v8f c) {
  return __builtin_amdgcn_wmma_f32_16x16x32_bf16(false, a, false, b, (short)0, c, false, false);
}

// CDNA5 async copy: 16B global -> LDS, tracked by ASYNCcnt (not LOADcnt).
__device__ inline void async_copy_b128(unsigned lds_off, const void* gptr) {
  asm volatile("global_load_async_to_lds_b128 %0, %1, off"
               :: "v"(lds_off), "v"(gptr) : "memory");
}
__device__ inline void wait_async0() {
  asm volatile("s_wait_asynccnt 0" ::: "memory");
}

// A fragment (16x32): lane -> M row, elements -> K (ISA 16-bit A table:
// lanes0-15 K=0..7,16..23; lanes16-31 K=8..15,24..31; K pairs contiguous).
__device__ inline v16bf load_fragA(const unsigned short* __restrict__ A, int lda,
                                   int m0, int k0) {
  int lane = threadIdx.x & 31;
  const unsigned short* p = A + (size_t)(m0 + (lane & 15)) * lda + k0 + ((lane >> 4) << 3);
  FragBF f;
#pragma unroll
  for (int j = 0; j < 8; ++j) {
    int kk = (j < 4) ? 2 * j : 2 * j + 8;
    f.u[j] = *(const unsigned*)(p + kk);
  }
  return f.v;
}

// B fragment (32x16): lane -> N column (lane&15), elements -> K (same table as A:
// lanes0-15 K=0..7,16..23; lanes16-31 K=8..15,24..31). B stored row-major [K][N].
__device__ inline v16bf load_fragB(const unsigned short* __restrict__ B, int ldn,
                                   int k0, int n0) {
  int lane = threadIdx.x & 31;
  const unsigned short* p = B + (size_t)(k0 + ((lane >> 4) << 3)) * ldn + n0 + (lane & 15);
  FragBF f;
#pragma unroll
  for (int j = 0; j < 16; ++j) f.h[j] = p[(size_t)((j < 8) ? j : j + 8) * ldn];
  return f.v;
}

// ---------------------------------------------------------------------------
// weight convert + transpose: dst[c*rows + r] = bf16(src[r*cols + c])
__global__ __launch_bounds__(256) void k_trans_f2bf(const float* __restrict__ src,
                                                    unsigned short* __restrict__ dst,
                                                    int rows, int cols) {
  int idx = blockIdx.x * 256 + threadIdx.x;
  if (idx >= rows * cols) return;
  int c = idx % cols, r = idx / cols;
  dst[(size_t)c * rows + r] = f2bf(src[idx]);
}

// ---------------------------------------------------------------------------
// LayerNorm: one row of 512 per block (256 thr x 2 elems), bf16 out
__global__ __launch_bounds__(256) void k_layernorm(const float* __restrict__ x,
                                                   const float* __restrict__ g,
                                                   const float* __restrict__ bb,
                                                   unsigned short* __restrict__ xn) {
  __shared__ float red[8];
  int row = blockIdx.x, tid = threadIdx.x;
  const float* xr = x + (size_t)row * 512;
  float v0 = xr[tid], v1 = xr[tid + 256];
  float s = v0 + v1;
#pragma unroll
  for (int o = 16; o > 0; o >>= 1) s += __shfl_xor(s, o, 32);
  if ((tid & 31) == 0) red[tid >> 5] = s;
  __syncthreads();
  float tot = 0.f;
#pragma unroll
  for (int i = 0; i < 8; ++i) tot += red[i];
  float mu = tot * (1.f / 512.f);
  float d0 = v0 - mu, d1 = v1 - mu;
  float q = d0 * d0 + d1 * d1;
#pragma unroll
  for (int o = 16; o > 0; o >>= 1) q += __shfl_xor(q, o, 32);
  __syncthreads();
  if ((tid & 31) == 0) red[tid >> 5] = q;
  __syncthreads();
  float vt = 0.f;
#pragma unroll
  for (int i = 0; i < 8; ++i) vt += red[i];
  float rstd = rsqrtf(vt * (1.f / 512.f) + 1e-5f);
  unsigned short* o = xn + (size_t)row * 512;
  o[tid]       = f2bf(d0 * rstd * g[tid] + bb[tid]);
  o[tid + 256] = f2bf(d1 * rstd * g[tid + 256] + bb[tid + 256]);
}

// ---------------------------------------------------------------------------
// 128x128 block-tile GEMM, async-LDS double-buffered pipeline.
// 8 waves (4M x 2N), wave tile 32x64, k-step 32.
// A tile 128x32 (8KB) + B tile 32x128 (8KB), x2 buffers = 32KB LDS.
__device__ inline void stage_tiles_async(const unsigned short* A, int lda, int m0,
                                         const unsigned short* B, int ldn, int n0,
                                         int k, unsigned short* As, unsigned short* Bs) {
  int t = threadIdx.x;
#pragma unroll
  for (int i = 0; i < 2; ++i) {          // A: 512 x 16B chunks
    int c = t + i * 256;
    int row = c >> 2, eo = (c & 3) * 8;  // element offset within 32-wide row
    async_copy_b128((unsigned)(size_t)&As[row * 32 + eo],
                    A + (size_t)(m0 + row) * lda + k + eo);
  }
#pragma unroll
  for (int i = 0; i < 2; ++i) {          // B: 512 x 16B chunks
    int c = t + i * 256;
    int kr = c >> 3, eo = (c & 7) * 8;   // element offset within 128-wide row
    async_copy_b128((unsigned)(size_t)&Bs[kr * 128 + eo],
                    B + (size_t)(k + kr) * ldn + n0 + eo);
  }
}

__device__ inline void gemm_block128_async(const unsigned short* A, int lda,
                                           const unsigned short* B, int ldn,
                                           int K, int m0blk, int n0blk,
                                           unsigned short* As0, unsigned short* As1,
                                           unsigned short* Bs0, unsigned short* Bs1,
                                           v8f acc[2][4], int wmL, int wnL) {
  stage_tiles_async(A, lda, m0blk, B, ldn, n0blk, 0, As0, Bs0);
  wait_async0();
  __syncthreads();
  int buf = 0;
  for (int k = 0; k < K; k += 32) {
    unsigned short* Ac = buf ? As1 : As0;
    unsigned short* Bc = buf ? Bs1 : Bs0;
    if (k + 32 < K)
      stage_tiles_async(A, lda, m0blk, B, ldn, n0blk, k + 32,
                        buf ? As0 : As1, buf ? Bs0 : Bs1);
    v16bf a0 = load_fragA(Ac, 32, wmL, 0);
    v16bf a1 = load_fragA(Ac, 32, wmL + 16, 0);
    v16bf b0 = load_fragB(Bc, 128, 0, wnL);
    v16bf b1 = load_fragB(Bc, 128, 0, wnL + 16);
    v16bf b2 = load_fragB(Bc, 128, 0, wnL + 32);
    v16bf b3 = load_fragB(Bc, 128, 0, wnL + 48);
    acc[0][0] = wmma_bf16(a0, b0, acc[0][0]);
    acc[0][1] = wmma_bf16(a0, b1, acc[0][1]);
    acc[0][2] = wmma_bf16(a0, b2, acc[0][2]);
    acc[0][3] = wmma_bf16(a0, b3, acc[0][3]);
    acc[1][0] = wmma_bf16(a1, b0, acc[1][0]);
    acc[1][1] = wmma_bf16(a1, b1, acc[1][1]);
    acc[1][2] = wmma_bf16(a1, b2, acc[1][2]);
    acc[1][3] = wmma_bf16(a1, b3, acc[1][3]);
    wait_async0();
    __syncthreads();
    buf ^= 1;
  }
}

// QKV GEMM: xn[16384,512] @ wqkvT -> scatter q(scaled), kT, v  (bf16)
__global__ __launch_bounds__(256) void k_gemm_qkv(const unsigned short* __restrict__ xn,
                                                  const unsigned short* __restrict__ wq,
                                                  unsigned short* __restrict__ q,
                                                  unsigned short* __restrict__ kT,
                                                  unsigned short* __restrict__ v) {
  __shared__ __align__(16) unsigned short As[2][128 * 32];
  __shared__ __align__(16) unsigned short Bs[2][32 * 128];
  int wid = threadIdx.x >> 5, lane = threadIdx.x & 31;
  int wmL = (wid >> 1) * 32, wnL = (wid & 1) * 64;
  int wm0 = blockIdx.x * 128 + wmL;
  int wn0 = blockIdx.y * 128 + wnL;
  v8f acc[2][4];
#pragma unroll
  for (int i = 0; i < 2; ++i)
#pragma unroll
    for (int j = 0; j < 4; ++j) acc[i][j] = vzero();
  gemm_block128_async(xn, 512, wq, 1536, 512, blockIdx.x * 128, blockIdx.y * 128,
                      As[0], As[1], Bs[0], Bs[1], acc, wmL, wnL);
#pragma unroll
  for (int i = 0; i < 2; ++i)
#pragma unroll
    for (int j = 0; j < 4; ++j)
#pragma unroll
      for (int r = 0; r < 8; ++r) {
        int gm = wm0 + i * 16 + r + ((lane >> 4) << 3);
        int gn = wn0 + j * 16 + (lane & 15);
        float val = acc[i][j][r];
        int b = gm >> 12, t = gm & 4095;
        if (gn < 512) {
          int h = gn >> 6, d = gn & 63;
          q[(((size_t)(b * 8 + h)) * 4096 + t) * 64 + d] = f2bf(val * 0.125f);
        } else if (gn < 1024) {
          int c = gn - 512; int h = c >> 6, d = c & 63;
          kT[(((size_t)(b * 8 + h)) * 64 + d) * 4096 + t] = f2bf(val);
        } else {
          int c = gn - 1024; int h = c >> 6, d = c & 63;
          v[(((size_t)(b * 8 + h)) * 4096 + t) * 64 + d] = f2bf(val);
        }
      }
}

// ---------------------------------------------------------------------------
// landmarks: ql[bh][j][d] = mean16(q); klT[bh][d][j] = mean16(kT)
__global__ __launch_bounds__(256) void k_landmark(const unsigned short* __restrict__ q,
                                                  const unsigned short* __restrict__ kT,
                                                  unsigned short* __restrict__ ql,
                                                  unsigned short* __restrict__ klT) {
  int idx = blockIdx.x * 256 + threadIdx.x;      // 32*256*64
  int d = idx & 63, j = (idx >> 6) & 255, bh = idx >> 14;
  const unsigned short* qp = q + (((size_t)bh * 4096) + j * 16) * 64 + d;
  float sq = 0.f;
#pragma unroll
  for (int t = 0; t < 16; ++t) sq += bf2f(qp[(size_t)t * 64]);
  ql[idx] = f2bf(sq * 0.0625f);
  const unsigned short* kp = kT + (((size_t)bh * 64) + d) * 4096 + j * 16;
  float sk = 0.f;
#pragma unroll
  for (int t = 0; t < 16; ++t) sk += bf2f(kp[t]);
  klT[(((size_t)bh * 64) + d) * 256 + j] = f2bf(sk * 0.0625f);
}

// ---------------------------------------------------------------------------
// sim + softmax: each wave computes 16 rows x 256 cols (K=64), row softmax.
// Used for a1 (Q=q, M=4096) and a2 (Q=ql, M=256, + column sums for pinv init).
__global__ __launch_bounds__(256) void k_sim_softmax(const unsigned short* __restrict__ Q,
                                                     const unsigned short* __restrict__ KlT,
                                                     unsigned short* __restrict__ out,
                                                     float* __restrict__ colsum, int Mrows) {
  int gw = (blockIdx.x * 256 + (int)threadIdx.x) >> 5;
  int wavesPerBh = Mrows >> 4;
  int bh = gw / wavesPerBh;
  int m0 = (gw - bh * wavesPerBh) << 4;
  int lane = threadIdx.x & 31;
  const unsigned short* Ab = Q + (size_t)bh * Mrows * 64;
  const unsigned short* Bb = KlT + (size_t)bh * 64 * 256;
  v16bf a0 = load_fragA(Ab, 64, m0, 0);
  v16bf a1 = load_fragA(Ab, 64, m0, 32);
  v8f c[16];
#pragma unroll
  for (int f = 0; f < 16; ++f) {
    v16bf b0 = load_fragB(Bb, 256, 0, f * 16);
    v16bf b1 = load_fragB(Bb, 256, 32, f * 16);
    c[f] = wmma_bf16(a1, b1, wmma_bf16(a0, b0, vzero()));
  }
#pragma unroll
  for (int r = 0; r < 8; ++r) {
    float mx = -1e30f;
#pragma unroll
    for (int f = 0; f < 16; ++f) mx = fmaxf(mx, c[f][r]);
#pragma unroll
    for (int o = 8; o > 0; o >>= 1) mx = fmaxf(mx, __shfl_xor(mx, o, 32));
    float sum = 0.f;
#pragma unroll
    for (int f = 0; f < 16; ++f) { float e = __expf(c[f][r] - mx); c[f][r] = e; sum += e; }
#pragma unroll
    for (int o = 8; o > 0; o >>= 1) sum += __shfl_xor(sum, o, 32);
    float inv = 1.f / sum;
#pragma unroll
    for (int f = 0; f < 16; ++f) c[f][r] *= inv;
  }
  unsigned short* Ob = out + ((size_t)bh * Mrows + m0) * 256;
  int hi = (lane >> 4) << 3;
#pragma unroll
  for (int f = 0; f < 16; ++f) {
    float cs = 0.f;
#pragma unroll
    for (int r = 0; r < 8; ++r) {
      Ob[(size_t)(r + hi) * 256 + f * 16 + (lane & 15)] = f2bf(c[f][r]);
      cs += c[f][r];
    }
    if (colsum) atomicAdd(&colsum[bh * 256 + f * 16 + (lane & 15)], cs);
  }
}

__global__ __launch_bounds__(256) void k_max_reduce(const float* __restrict__ colsum,
                                                    float* __restrict__ s) {
  __shared__ float red[8];
  float m = 0.f;
  for (int i = threadIdx.x; i < 32 * 256; i += 256) m = fmaxf(m, colsum[i]);
#pragma unroll
  for (int o = 16; o > 0; o >>= 1) m = fmaxf(m, __shfl_xor(m, o, 32));
  if ((threadIdx.x & 31) == 0) red[threadIdx.x >> 5] = m;
  __syncthreads();
  if (threadIdx.x == 0) {
    float t = red[0];
    for (int i = 1; i < 8; ++i) t = fmaxf(t, red[i]);
    s[0] = t;                       // row-sum max of softmax ~ 1.0
  }
}

// z0 = a2^T / s
__global__ __launch_bounds__(256) void k_zinit(const unsigned short* __restrict__ a2,
                                               const float* __restrict__ s,
                                               unsigned short* __restrict__ z) {
  size_t idx = (size_t)blockIdx.x * 256 + threadIdx.x;    // 32*256*256
  int j = (int)(idx & 255), i = (int)((idx >> 8) & 255), bh = (int)(idx >> 16);
  float inv = 1.f / s[0];
  float v = bf2f(a2[((size_t)bh << 16) + ((size_t)j << 8) + i]) * inv;
  z[idx] = f2bf(v);
}

// ---------------------------------------------------------------------------
// pinv GEMM (batched 32x): O = alpha * A @ B', B' = bneg ? (bdiag*I - B) : B.
// Wave tile 16x32; per bh 16x8 tiles; diagonal test from fragment (k,n) coords.
__global__ __launch_bounds__(256) void k_pinv_gemm(const unsigned short* __restrict__ A,
                                                   const unsigned short* __restrict__ B,
                                                   unsigned short* __restrict__ O,
                                                   float alpha, float bdiag, int bneg) {
  int gw = (blockIdx.x * 256 + (int)threadIdx.x) >> 5;    // 4096 waves
  int bh = gw >> 7, t = gw & 127;
  int m0 = (t >> 3) << 4, n0 = (t & 7) << 5;
  const unsigned short* Ab = A + ((size_t)bh << 16);
  const unsigned short* Bb = B + ((size_t)bh << 16);
  int lane = threadIdx.x & 31;
  int klane = (lane >> 4) << 3;
  v8f c0 = vzero(), c1 = vzero();
  for (int k = 0; k < 256; k += 32) {
    v16bf a = load_fragA(Ab, 256, m0, k);
    FragBF b0f, b1f;
    b0f.v = load_fragB(Bb, 256, k, n0);
    b1f.v = load_fragB(Bb, 256, k, n0 + 16);
    if (bneg) {
      int gk = k + klane;
#pragma unroll
      for (int e = 0; e < 16; ++e) {
        int ke = gk + ((e < 8) ? e : e + 8);
        b0f.h[e] = f2bf(((ke == n0 + (lane & 15)) ? bdiag : 0.f) - bf2f(b0f.h[e]));
        b1f.h[e] = f2bf(((ke == n0 + 16 + (lane & 15)) ? bdiag : 0.f) - bf2f(b1f.h[e]));
      }
    }
    c0 = wmma_bf16(a, b0f.v, c0);
    c1 = wmma_bf16(a, b1f.v, c1);
  }
  unsigned short* Ob = O + ((size_t)bh << 16);
#pragma unroll
  for (int r = 0; r < 8; ++r) {
    int gm = m0 + r + ((lane >> 4) << 3);
    int gn = n0 + (lane & 15);
    Ob[(size_t)gm * 256 + gn]      = f2bf(alpha * c0[r]);
    Ob[(size_t)gm * 256 + gn + 16] = f2bf(alpha * c1[r]);
  }
}

// ---------------------------------------------------------------------------
// flash-fused av = softmax(ql @ k^T) @ v : per wave 16 landmark rows, loop over
// 32-key blocks, P staged via LDS to reshape C-layout -> A-layout.
__global__ __launch_bounds__(256) void k_flash_a3v(const unsigned short* __restrict__ ql,
                                                   const unsigned short* __restrict__ kT,
                                                   const unsigned short* __restrict__ v,
                                                   unsigned short* __restrict__ av) {
  __shared__ __align__(16) unsigned short ptile[8][16][32];
  int wid = threadIdx.x >> 5, lane = threadIdx.x & 31;
  int bh = blockIdx.x >> 1;
  int m0 = ((blockIdx.x & 1) * 8 + wid) * 16;
  const unsigned short* Ab = ql + (size_t)bh * 256 * 64;
  const unsigned short* Kb = kT + (size_t)bh * 64 * 4096;
  const unsigned short* Vb = v + (size_t)bh * 4096 * 64;
  v16bf a0 = load_fragA(Ab, 64, m0, 0);
  v16bf a1 = load_fragA(Ab, 64, m0, 32);
  v8f o0 = vzero(), o1 = vzero(), o2 = vzero(), o3 = vzero();
  float mrow[8], lrow[8];
#pragma unroll
  for (int r = 0; r < 8; ++r) { mrow[r] = -1e30f; lrow[r] = 0.f; }
  for (int kb = 0; kb < 4096; kb += 32) {
    v8f s0 = wmma_bf16(a1, load_fragB(Kb, 4096, 32, kb),
             wmma_bf16(a0, load_fragB(Kb, 4096, 0, kb), vzero()));
    v8f s1 = wmma_bf16(a1, load_fragB(Kb, 4096, 32, kb + 16),
             wmma_bf16(a0, load_fragB(Kb, 4096, 0, kb + 16), vzero()));
#pragma unroll
    for (int r = 0; r < 8; ++r) {
      float mx = fmaxf(s0[r], s1[r]);
#pragma unroll
      for (int o = 8; o > 0; o >>= 1) mx = fmaxf(mx, __shfl_xor(mx, o, 32));
      float mnew = fmaxf(mrow[r], mx);
      float scale = __expf(mrow[r] - mnew);
      float e0 = __expf(s0[r] - mnew), e1 = __expf(s1[r] - mnew);
      s0[r] = e0; s1[r] = e1;
      float ls = e0 + e1;
#pragma unroll
      for (int o = 8; o > 0; o >>= 1) ls += __shfl_xor(ls, o, 32);
      lrow[r] = lrow[r] * scale + ls;
      mrow[r] = mnew;
      o0[r] *= scale; o1[r] *= scale; o2[r] *= scale; o3[r] *= scale;
    }
    __syncthreads();
    int hi = (lane >> 4) << 3;
#pragma unroll
    for (int r = 0; r < 8; ++r) {
      ptile[wid][r + hi][lane & 15]        = f2bf(s0[r]);
      ptile[wid][r + hi][16 + (lane & 15)] = f2bf(s1[r]);
    }
    __syncthreads();
    FragBF pf;
    const unsigned short* prow = &ptile[wid][lane & 15][(lane >> 4) << 3];
#pragma unroll
    for (int j = 0; j < 8; ++j) {
      int kk = (j < 4) ? 2 * j : 2 * j + 8;
      pf.u[j] = *(const unsigned*)(prow + kk);
    }
    o0 = wmma_bf16(pf.v, load_fragB(Vb, 64, kb, 0),  o0);
    o1 = wmma_bf16(pf.v, load_fragB(Vb, 64, kb, 16), o1);
    o2 = wmma_bf16(pf.v, load_fragB(Vb, 64, kb, 32), o2);
    o3 = wmma_bf16(pf.v, load_fragB(Vb, 64, kb, 48), o3);
  }
#pragma unroll
  for (int r = 0; r < 8; ++r) {
    float inv = 1.f / lrow[r];
    int j = m0 + r + ((lane >> 4) << 3);
    unsigned short* op = av + ((size_t)bh * 256 + j) * 64 + (lane & 15);
    op[0]  = f2bf(o0[r] * inv);
    op[16] = f2bf(o1[r] * inv);
    op[32] = f2bf(o2[r] * inv);
    op[48] = f2bf(o3[r] * inv);
  }
}

// ---------------------------------------------------------------------------
// W = z_final @ av : per bh 256x64, K=256 (wave tile 16x32)
__global__ __launch_bounds__(256) void k_w(const unsigned short* __restrict__ z,
                                           const unsigned short* __restrict__ av,
                                           unsigned short* __restrict__ w) {
  int gw = (blockIdx.x * 256 + (int)threadIdx.x) >> 5;    // 1024 waves
  int bh = gw >> 5, t = gw & 31;
  int m0 = (t >> 1) << 4, n0 = (t & 1) << 5;
  const unsigned short* Ab = z + ((size_t)bh << 16);
  const unsigned short* Bb = av + (size_t)bh * 256 * 64;
  v8f c0 = vzero(), c1 = vzero();
  for (int k = 0; k < 256; k += 32) {
    v16bf a = load_fragA(Ab, 256, m0, k);
    c0 = wmma_bf16(a, load_fragB(Bb, 64, k, n0), c0);
    c1 = wmma_bf16(a, load_fragB(Bb, 64, k, n0 + 16), c1);
  }
  int lane = threadIdx.x & 31;
  unsigned short* Ob = w + (size_t)bh * 256 * 64;
#pragma unroll
  for (int r = 0; r < 8; ++r) {
    int gm = m0 + r + ((lane >> 4) << 3);
    int gn = n0 + (lane & 15);
    Ob[(size_t)gm * 64 + gn]      = f2bf(c0[r]);
    Ob[(size_t)gm * 64 + gn + 16] = f2bf(c1[r]);
  }
}

// depthwise conv residual on v (kernel 33 along tokens), writes f32 [b,n,512]
__global__ __launch_bounds__(256) void k_conv(const unsigned short* __restrict__ v,
                                              const float* __restrict__ cw,
                                              float* __restrict__ res) {
  size_t idx = (size_t)blockIdx.x * 256 + threadIdx.x;    // 4*512*4096
  int n = (int)(idx & 4095);
  int c = (int)((idx >> 12) & 511);
  int b = (int)(idx >> 21);
  int h = c >> 6, d = c & 63;
  const unsigned short* vrow = v + (((size_t)(b * 8 + h)) * 4096) * 64 + d;
  float acc = 0.f;
#pragma unroll
  for (int t = 0; t < 33; ++t) {
    int p = n + t - 16;
    if (p >= 0 && p < 4096) acc += cw[h * 33 + t] * bf2f(vrow[(size_t)p * 64]);
  }
  res[((size_t)b * 4096 + n) * 512 + c] = acc;
}

// out = a1 @ W + conv_res, merged heads, bf16 xo[b,n,512]
__global__ __launch_bounds__(256) void k_outattn(const unsigned short* __restrict__ a1,
                                                 const unsigned short* __restrict__ w,
                                                 const float* __restrict__ res,
                                                 unsigned short* __restrict__ xo) {
  int gw = (blockIdx.x * 256 + (int)threadIdx.x) >> 5;    // 16384 waves
  int bh = gw >> 9, t = gw & 511;
  int m0 = (t >> 1) << 4, n0 = (t & 1) << 5;
  const unsigned short* Ab = a1 + (size_t)bh * 4096 * 256;
  const unsigned short* Bb = w + (size_t)bh * 256 * 64;
  v8f c0 = vzero(), c1 = vzero();
  for (int k = 0; k < 256; k += 32) {
    v16bf a = load_fragA(Ab, 256, m0, k);
    c0 = wmma_bf16(a, load_fragB(Bb, 64, k, n0), c0);
    c1 = wmma_bf16(a, load_fragB(Bb, 64, k, n0 + 16), c1);
  }
  int lane = threadIdx.x & 31;
  int b = bh >> 3, h = bh & 7;
#pragma unroll
  for (int r = 0; r < 8; ++r) {
    int tok = m0 + r + ((lane >> 4) << 3);
    int d = n0 + (lane & 15);
    size_t oi = ((size_t)b * 4096 + tok) * 512 + h * 64 + d;
    xo[oi]      = f2bf(c0[r] + res[oi]);
    xo[oi + 16] = f2bf(c1[r] + res[oi + 16]);
  }
}

// final: y = xo @ w_out^T + b_out + x*omega   (f32 out), async-LDS GEMM
__global__ __launch_bounds__(256) void k_final(const unsigned short* __restrict__ xo,
                                               const unsigned short* __restrict__ woT,
                                               const float* __restrict__ bout,
                                               const float* __restrict__ x,
                                               const float* __restrict__ omega,
                                               float* __restrict__ y) {
  __shared__ __align__(16) unsigned short As[2][128 * 32];
  __shared__ __align__(16) unsigned short Bs[2][32 * 128];
  int wid = threadIdx.x >> 5, lane = threadIdx.x & 31;
  int wmL = (wid >> 1) * 32, wnL = (wid & 1) * 64;
  int wm0 = blockIdx.x * 128 + wmL;
  int wn0 = blockIdx.y * 128 + wnL;
  v8f acc[2][4];
#pragma unroll
  for (int i = 0; i < 2; ++i)
#pragma unroll
    for (int j = 0; j < 4; ++j) acc[i][j] = vzero();
  gemm_block128_async(xo, 512, woT, 512, 512, blockIdx.x * 128, blockIdx.y * 128,
                      As[0], As[1], Bs[0], Bs[1], acc, wmL, wnL);
  float om = omega[0];
#pragma unroll
  for (int i = 0; i < 2; ++i)
#pragma unroll
    for (int j = 0; j < 4; ++j)
#pragma unroll
      for (int r = 0; r < 8; ++r) {
        int gm = wm0 + i * 16 + r + ((lane >> 4) << 3);
        int gn = wn0 + j * 16 + (lane & 15);
        size_t oi = (size_t)gm * 512 + gn;
        y[oi] = acc[i][j][r] + bout[gn] + x[oi] * om;
      }
}

// ---------------------------------------------------------------------------
extern "C" void kernel_launch(void* const* d_in, const int* in_sizes, int n_in,
                              void* d_out, int out_size, void* d_ws, size_t ws_size,
                              hipStream_t stream) {
  (void)in_sizes; (void)n_in; (void)out_size; (void)ws_size;
  const float* x      = (const float*)d_in[0];
  const float* ln_g   = (const float*)d_in[1];
  const float* ln_b   = (const float*)d_in[2];
  const float* w_qkv  = (const float*)d_in[3];
  const float* w_out  = (const float*)d_in[4];
  const float* b_out  = (const float*)d_in[5];
  const float* conv_w = (const float*)d_in[6];
  const float* omega  = (const float*)d_in[7];
  float* y = (float*)d_out;

  char* base = (char*)d_ws; size_t off = 0;
  auto alloc = [&](size_t bytes) -> void* {
    void* r = base + off; off += (bytes + 255) & ~(size_t)255; return r;
  };
  // ~216 MB total workspace
  unsigned short* xnb    = (unsigned short*)alloc((size_t)16384 * 512 * 2);
  unsigned short* wqkvT  = (unsigned short*)alloc((size_t)512 * 1536 * 2);
  unsigned short* woutT  = (unsigned short*)alloc((size_t)512 * 512 * 2);
  unsigned short* qb     = (unsigned short*)alloc((size_t)32 * 4096 * 64 * 2);
  unsigned short* kTb    = (unsigned short*)alloc((size_t)32 * 64 * 4096 * 2);
  unsigned short* vb     = (unsigned short*)alloc((size_t)32 * 4096 * 64 * 2);
  unsigned short* qlb    = (unsigned short*)alloc((size_t)32 * 256 * 64 * 2);
  unsigned short* klTb   = (unsigned short*)alloc((size_t)32 * 64 * 256 * 2);
  unsigned short* a1b    = (unsigned short*)alloc((size_t)32 * 4096 * 256 * 2);
  unsigned short* a2b    = (unsigned short*)alloc((size_t)32 * 256 * 256 * 2);
  float*          colsum = (float*)alloc((size_t)32 * 256 * 4);
  float*          sval   = (float*)alloc(256);
  unsigned short* zA     = (unsigned short*)alloc((size_t)32 * 256 * 256 * 2);
  unsigned short* zB     = (unsigned short*)alloc((size_t)32 * 256 * 256 * 2);
  unsigned short* xzb    = (unsigned short*)alloc((size_t)32 * 256 * 256 * 2);
  unsigned short* m2b    = (unsigned short*)alloc((size_t)32 * 256 * 256 * 2);
  unsigned short* m3b    = (unsigned short*)alloc((size_t)32 * 256 * 256 * 2);
  unsigned short* avb    = (unsigned short*)alloc((size_t)32 * 256 * 64 * 2);
  unsigned short* wb     = (unsigned short*)alloc((size_t)32 * 256 * 64 * 2);
  float*          resf   = (float*)alloc((size_t)16384 * 512 * 4);
  unsigned short* xob    = (unsigned short*)alloc((size_t)16384 * 512 * 2);

  k_trans_f2bf<<<3072, 256, 0, stream>>>(w_qkv, wqkvT, 1536, 512);
  k_trans_f2bf<<<1024, 256, 0, stream>>>(w_out, woutT, 512, 512);
  k_layernorm<<<16384, 256, 0, stream>>>(x, ln_g, ln_b, xnb);
  k_gemm_qkv<<<dim3(128, 12), 256, 0, stream>>>(xnb, wqkvT, qb, kTb, vb);
  k_landmark<<<2048, 256, 0, stream>>>(qb, kTb, qlb, klTb);
  k_sim_softmax<<<1024, 256, 0, stream>>>(qb, klTb, a1b, nullptr, 4096);
  hipMemsetAsync(colsum, 0, (size_t)32 * 256 * 4, stream);
  k_sim_softmax<<<64, 256, 0, stream>>>(qlb, klTb, a2b, colsum, 256);
  k_max_reduce<<<1, 256, 0, stream>>>(colsum, sval);
  k_zinit<<<8192, 256, 0, stream>>>(a2b, sval, zA);

  unsigned short* zc = zA; unsigned short* zn = zB;
  for (int it = 0; it < 6; ++it) {
    k_pinv_gemm<<<512, 256, 0, stream>>>(a2b, zc, xzb, 1.f, 0.f, 0);   // xz = a2@z
    k_pinv_gemm<<<512, 256, 0, stream>>>(xzb, xzb, m2b, 1.f, 7.f, 1);  // xz@(7I-xz)
    k_pinv_gemm<<<512, 256, 0, stream>>>(xzb, m2b, m3b, 1.f, 15.f, 1); // xz@(15I-m2)
    k_pinv_gemm<<<512, 256, 0, stream>>>(zc, m3b, zn, 0.25f, 13.f, 1); // z@(13I-m3)/4
    unsigned short* t = zc; zc = zn; zn = t;
  }

  k_flash_a3v<<<64, 256, 0, stream>>>(qlb, kTb, vb, avb);
  k_w<<<128, 256, 0, stream>>>(zc, avb, wb);
  k_conv<<<32768, 256, 0, stream>>>(vb, conv_w, resf);
  k_outattn<<<2048, 256, 0, stream>>>(a1b, wb, resf, xob);
  k_final<<<dim3(128, 4), 256, 0, stream>>>(xob, woutT, b_out, x, omega, y);
}
